// CausalAttention_50878182589030
// MI455X (gfx1250) — compile-verified, matrix operand order checked by reading, not run on hardware
//
#include <hip/hip_runtime.h>
#include <hip/hip_bf16.h>

typedef __bf16 bf16;
typedef __attribute__((ext_vector_type(16))) bf16  v16bf;
typedef __attribute__((ext_vector_type(8)))  bf16  v8bf;
typedef __attribute__((ext_vector_type(8)))  float v8f;

__device__ __forceinline__ v8bf ld8(const bf16* p) {
    return *reinterpret_cast<const v8bf*>(p);
}
__device__ __forceinline__ v16bf combine(v8bf lo, v8bf hi) {
    return __builtin_shufflevector(lo, hi, 0,1,2,3,4,5,6,7,8,9,10,11,12,13,14,15);
}
__device__ __forceinline__ v8f wmma_bf16(v16bf a, v16bf b, v8f c) {
    return __builtin_amdgcn_wmma_f32_16x16x32_bf16(false, a, false, b, (short)0, c, false, false);
}
// 16-byte global -> LDS async copy (ASYNCcnt-tracked, no VGPR data path).
// LDS operand = workgroup-relative byte offset = low 32 bits of generic pointer.
__device__ __forceinline__ void async16(const bf16* g, bf16* l) {
    asm volatile("global_load_async_to_lds_b128 %0, %1, off"
                 :: "v"((unsigned)(unsigned long long)l), "v"(g)
                 : "memory");
}
__device__ __forceinline__ void wait_async() {
    asm volatile("s_wait_asynccnt 0" ::: "memory");
}
__device__ __forceinline__ void wait_ds() {
    asm volatile("s_wait_dscnt 0" ::: "memory");
}

// ---------------------------------------------------------------- converts
__global__ void f32_to_bf16_kernel(const float* __restrict__ in,
                                   bf16* __restrict__ out, int n) {
    for (int i = blockIdx.x * blockDim.x + threadIdx.x; i < n;
         i += gridDim.x * blockDim.x)
        out[i] = (bf16)in[i];
}
// out[c][r] = (bf16) in[r][c]   (weights: store B^T so GEMM staging is copy-shaped)
__global__ void f32_to_bf16_tr_kernel(const float* __restrict__ in,
                                      bf16* __restrict__ out, int rows, int cols) {
    int n = rows * cols;
    for (int i = blockIdx.x * blockDim.x + threadIdx.x; i < n;
         i += gridDim.x * blockDim.x) {
        int r = i / cols, c = i % cols;
        out[(size_t)c * rows + r] = (bf16)in[i];
    }
}

// ---------------------------------------------------------------- GEMM
// A:  [4096, 1024] bf16 row-major.  Bt: [Nout, 1024] bf16 (pre-transposed).
// Block tile 128x128, 8 waves (4M x 2N), wave tile 32x64 -> 8 WMMA / K-step.
// mode 0: QKV epilogue (Q scaled, K, V transposed; bf16). mode 1: proj + bias, f32.
__global__ __launch_bounds__(256) void gemm_bf16_kernel(
    const bf16* __restrict__ A, const bf16* __restrict__ Bt, int mode,
    bf16* __restrict__ Qout, bf16* __restrict__ Kout, bf16* __restrict__ Vtout,
    float* __restrict__ Cout, const float* __restrict__ bias)
{
    __shared__ bf16 sA[128 * 40];   // stride 40 -> conflict-free b128 frag reads
    __shared__ bf16 sB[128 * 40];

    const int tid  = threadIdx.x;
    const int wid  = tid >> 5;
    const int lane = tid & 31;
    const int half = lane >> 4, lm = lane & 15, h8 = half * 8;
    const int waveM = wid & 3, waveN = wid >> 2;
    const int mBase = blockIdx.y * 128;
    const int nBase = blockIdx.x * 128;

    v8f acc[2][4];
#pragma unroll
    for (int mt = 0; mt < 2; ++mt)
#pragma unroll
        for (int nt = 0; nt < 4; ++nt) acc[mt][nt] = {};

    const int sr = tid >> 1, sc = (tid & 1) * 16;   // staging: 16 elems / thread / tile

    for (int kk = 0; kk < 1024; kk += 32) {
        const bf16* ga = A  + (size_t)(mBase + sr) * 1024 + kk + sc;
        const bf16* gb = Bt + (size_t)(nBase + sr) * 1024 + kk + sc;
        async16(ga,     &sA[sr * 40 + sc]);
        async16(ga + 8, &sA[sr * 40 + sc + 8]);
        async16(gb,     &sB[sr * 40 + sc]);
        async16(gb + 8, &sB[sr * 40 + sc + 8]);
        if (kk + 32 < 1024) {
            __builtin_prefetch(ga + 32, 0, 1);
            __builtin_prefetch(gb + 32, 0, 1);
        }
        wait_async();
        __syncthreads();

        v16bf aF[2], bF[4];
#pragma unroll
        for (int mt = 0; mt < 2; ++mt) {
            int r = waveM * 32 + mt * 16 + lm;
            aF[mt] = combine(ld8(&sA[r * 40 + h8]), ld8(&sA[r * 40 + 16 + h8]));
        }
#pragma unroll
        for (int nt = 0; nt < 4; ++nt) {
            int c = waveN * 64 + nt * 16 + lm;
            bF[nt] = combine(ld8(&sB[c * 40 + h8]), ld8(&sB[c * 40 + 16 + h8]));
        }
#pragma unroll
        for (int mt = 0; mt < 2; ++mt)
#pragma unroll
            for (int nt = 0; nt < 4; ++nt)
                acc[mt][nt] = wmma_bf16(aF[mt], bF[nt], acc[mt][nt]);
        __syncthreads();
    }

    // epilogue; C/D layout: element (lane, v) -> row v+8*half, col lm
#pragma unroll
    for (int mt = 0; mt < 2; ++mt)
#pragma unroll
        for (int nt = 0; nt < 4; ++nt)
#pragma unroll
            for (int v = 0; v < 8; ++v) {
                int row = mBase + waveM * 32 + mt * 16 + v + 8 * half;
                int col = nBase + waveN * 64 + nt * 16 + lm;
                float val = acc[mt][nt][v];
                if (mode == 0) {
                    int bb = row >> 11, ns = row & 2047;      // B*N = 4096
                    int which = col >> 10, rem = col & 1023;  // q/k/v chunks of 1024
                    int hh = rem >> 6, d = rem & 63;
                    int bh = bb * 16 + hh;
                    if (which == 0)
                        Qout[((size_t)bh * 2048 + ns) * 64 + d] = (bf16)(val * 0.125f);
                    else if (which == 1)
                        Kout[((size_t)bh * 2048 + ns) * 64 + d] = (bf16)val;
                    else
                        Vtout[((size_t)bh * 64 + d) * 2048 + ns] = (bf16)val;
                } else {
                    Cout[(size_t)row * 1024 + col] = val + bias[col];
                }
            }
}

// ---------------------------------------------------------------- flash attention
// One block (8 waves) per (b, h, 128-query chunk); wave w owns queries
// [qc*128 + w*16, +16). K/V 32-key tiles staged once per block via async-to-LDS.
// Q/K: [B,H,N,64] bf16 (Q pre-scaled). Vt: [B,H,64,N] bf16. Out: [B,N,C] bf16.
__global__ __launch_bounds__(256) void flash_attn_kernel(
    const bf16* __restrict__ Q, const bf16* __restrict__ K,
    const bf16* __restrict__ Vt, bf16* __restrict__ Obf)
{
    __shared__ bf16 sK[32 * 72];        // [key][d], stride 72: conflict-free
    __shared__ bf16 sV[64 * 40];        // [d][key], stride 40
    __shared__ bf16 pbuf[8][16 * 40];   // per-wave P tile (16q x 32k)

    const int tid  = threadIdx.x;
    const int widb = tid >> 5;
    const int lane = tid & 31;
    const int half = lane >> 4, lm = lane & 15, h8 = half * 8;

    const int b   = blockIdx.x >> 8;            // 16 h * 16 chunks = 256 / batch
    const int rm  = blockIdx.x & 255;
    const int h   = rm >> 4;
    const int qcb = (rm & 15) << 7;             // block query base
    const int qbase = qcb + widb * 16;          // wave query base
    const int qEnd  = qbase + 15;

    bf16* pw = &pbuf[widb][0];
    const bf16* Qbh = Q  + (size_t)(b * 16 + h) * 2048 * 64;
    const bf16* Kbh = K  + (size_t)(b * 16 + h) * 2048 * 64;
    const bf16* Vbh = Vt + (size_t)(b * 16 + h) * 64 * 2048;

    const bf16* qr = Qbh + (size_t)(qbase + lm) * 64;
    const v16bf qA0 = combine(ld8(qr + h8),      ld8(qr + 16 + h8));
    const v16bf qA1 = combine(ld8(qr + 32 + h8), ld8(qr + 48 + h8));

    v8f o[4]; o[0] = {}; o[1] = {}; o[2] = {}; o[3] = {};
    float m8[8], l8[8];
    const float INF = __builtin_inff();
#pragma unroll
    for (int v = 0; v < 8; ++v) { m8[v] = -INF; l8[v] = 0.f; }

    const int kk0 = tid >> 3, kc0 = (tid & 7) * 8;   // K staging slot (32x64)
    const int vd0 = tid >> 2, vk0 = (tid & 3) * 8;   // V staging slot (64x32)

    for (int j = 0; j <= qcb + 127; j += 32) {
        // ---- stage K tile [j..j+32) x 64d and V tile 64d x [j..j+32) once per block
        async16(Kbh + (size_t)(j + kk0) * 64 + kc0, &sK[kk0 * 72 + kc0]);
        async16(Vbh + (size_t)vd0 * 2048 + j + vk0, &sV[vd0 * 40 + vk0]);
        wait_async();
        __syncthreads();

        if (j <= qEnd) {   // wave-uniform: this wave still needs these keys
            // scores: two 16x16 tiles over the 32 staged keys
            v8f s0 = {}, s1 = {};
            {
                const bf16* kr0 = &sK[lm * 72];
                const bf16* kr1 = &sK[(16 + lm) * 72];
                s0 = wmma_bf16(qA0, combine(ld8(kr0 + h8),      ld8(kr0 + 16 + h8)), s0);
                s0 = wmma_bf16(qA1, combine(ld8(kr0 + 32 + h8), ld8(kr0 + 48 + h8)), s0);
                s1 = wmma_bf16(qA0, combine(ld8(kr1 + h8),      ld8(kr1 + 16 + h8)), s1);
                s1 = wmma_bf16(qA1, combine(ld8(kr1 + 32 + h8), ld8(kr1 + 48 + h8)), s1);
            }
            const bool need_mask = (j + 31 > qbase);

            // online softmax per output row (row = v + 8*half)
#pragma unroll
            for (int v = 0; v < 8; ++v) {
                float a = s0[v], c = s1[v];
                if (need_mask) {
                    int qrow = qbase + v + 8 * half;
                    if (j + lm      > qrow) a = -INF;
                    if (j + 16 + lm > qrow) c = -INF;
                }
                float r = fmaxf(a, c);
                r = fmaxf(r, __shfl_xor(r, 1));
                r = fmaxf(r, __shfl_xor(r, 2));
                r = fmaxf(r, __shfl_xor(r, 4));
                r = fmaxf(r, __shfl_xor(r, 8));
                float mn = fmaxf(m8[v], r);
                float sc = __expf(m8[v] - mn);
                float e0 = __expf(a - mn);
                float e1 = __expf(c - mn);
                float rs = e0 + e1;
                rs += __shfl_xor(rs, 1);
                rs += __shfl_xor(rs, 2);
                rs += __shfl_xor(rs, 4);
                rs += __shfl_xor(rs, 8);
                l8[v] = l8[v] * sc + rs;
                m8[v] = mn;
                o[0][v] *= sc; o[1][v] *= sc; o[2][v] *= sc; o[3][v] *= sc;
                int pr = v + 8 * half;
                pw[pr * 40 + lm]      = (bf16)e0;
                pw[pr * 40 + 16 + lm] = (bf16)e1;
            }
            wait_ds();   // cross-lane LDS round trip within the wave

            v16bf pA = combine(ld8(&pw[lm * 40 + h8]), ld8(&pw[lm * 40 + 16 + h8]));
#pragma unroll
            for (int t = 0; t < 4; ++t) {
                const bf16* vr = &sV[(t * 16 + lm) * 40];
                o[t] = wmma_bf16(pA, combine(ld8(vr + h8), ld8(vr + 16 + h8)), o[t]);
            }
        }
        __syncthreads();   // protect sK/sV before next stage overwrites
    }

    // ---- normalize, store bf16 [B, N, C] for the proj GEMM
    bf16* orow = Obf + (size_t)(b * 2048 + qbase) * 1024 + h * 64;
#pragma unroll
    for (int t = 0; t < 4; ++t)
#pragma unroll
        for (int v = 0; v < 8; ++v) {
            int rr = v + 8 * half;
            orow[(size_t)rr * 1024 + t * 16 + lm] = (bf16)(o[t][v] / l8[v]);
        }
}

// ---------------------------------------------------------------- launch
extern "C" void kernel_launch(void* const* d_in, const int* in_sizes, int n_in,
                              void* d_out, int out_size, void* d_ws, size_t ws_size,
                              hipStream_t stream) {
    const float* x      = (const float*)d_in[0];   // [2,2048,1024]
    const float* w_qkv  = (const float*)d_in[1];   // [1024,3072]
    const float* w_proj = (const float*)d_in[2];   // [1024,1024]
    const float* b_proj = (const float*)d_in[3];   // [1024]
    float* out = (float*)d_out;                    // [2,2048,1024]

    char* ws = (char*)d_ws;
    bf16* xbf    = (bf16*)(ws);                    //  8 MB  [4096,1024]
    bf16* wqkvT  = (bf16*)(ws + 8388608);          //  6 MB  [3072,1024] (B^T)
    bf16* wprojT = (bf16*)(ws + 14680064);         //  2 MB  [1024,1024] (B^T)
    bf16* Qbf    = (bf16*)(ws + 16777216);         //  8 MB  [B,H,N,64]
    bf16* Kbf    = (bf16*)(ws + 25165824);         //  8 MB  [B,H,N,64]
    bf16* Vtbf   = (bf16*)(ws + 33554432);         //  8 MB  [B,H,64,N]
    bf16* attnbf = (bf16*)(ws + 41943040);         //  8 MB  [4096,1024]

    f32_to_bf16_kernel<<<2048, 256, 0, stream>>>(x, xbf, 4096 * 1024);
    f32_to_bf16_tr_kernel<<<2048, 256, 0, stream>>>(w_qkv,  wqkvT,  1024, 3072);
    f32_to_bf16_tr_kernel<<<1024, 256, 0, stream>>>(w_proj, wprojT, 1024, 1024);

    // x @ w_qkv -> scatter Q(scaled)/K/V^T
    gemm_bf16_kernel<<<dim3(24, 32), 256, 0, stream>>>(
        xbf, wqkvT, 0, Qbf, Kbf, Vtbf, nullptr, nullptr);

    // causal flash attention: block = 8 waves sharing K/V tiles via LDS
    flash_attn_kernel<<<512, 256, 0, stream>>>(Qbf, Kbf, Vtbf, attnbf);

    // attn @ w_proj + bias -> f32 out
    gemm_bf16_kernel<<<dim3(8, 32), 256, 0, stream>>>(
        attnbf, wprojT, 1, nullptr, nullptr, nullptr, out, b_proj);
}
